// SpikingUnifiedCoreFlow_28020366639241
// MI455X (gfx1250) — matrix operand order repaired and checked by reading, not compiled, not run on hardware
//
#include <hip/hip_runtime.h>

typedef __attribute__((ext_vector_type(16))) _Float16 v16h;
typedef __attribute__((ext_vector_type(8)))  _Float16 v8h;
typedef __attribute__((ext_vector_type(8)))  float    v8f;
typedef int v4i __attribute__((vector_size(16)));   // matches builtin param type

// T=32, B=128, D_in=2048, D_hid=4096, D_out=2048
#define SIM_T 32
#define BATCH 128
#define D_IN  2048
#define D_HID 4096
#define D_OUT 2048

#if defined(__AMDGCN__) && __has_builtin(__builtin_amdgcn_global_load_async_to_lds_b128) && \
    __has_builtin(__builtin_amdgcn_s_wait_asynccnt)
#define USE_ASYNC 1
#else
#define USE_ASYNC 0
#endif

// ---------------- fp32 -> fp16 weight convert ----------------
__global__ __launch_bounds__(256) void convert_w_kernel(const float* __restrict__ w,
                                                        _Float16* __restrict__ wh,
                                                        int n) {
  int i = blockIdx.x * 256 + threadIdx.x;
  if (i < n) wh[i] = (_Float16)w[i];
}

// ---------------- uniform spike encoder ----------------
__global__ __launch_bounds__(256) void encode_kernel(const float* __restrict__ x,
                                                     _Float16* __restrict__ tr) {
  const int idx = blockIdx.x * 256 + threadIdx.x;   // over B*D_IN = 262144
  const float v = x[idx];
  const int   N = (int)rintf(v * 32.0f);            // round-half-even like jnp.round
  const bool  full = (N == SIM_T);
  const bool  mask = (N != 0) && !full;
  const int   Ns = (N > 1) ? N : 1;
  const float spacing = 32.0f / (float)Ns;
  #pragma unroll
  for (int t = 0; t < SIM_T; ++t) {
    const float c = (float)t;
    const bool res = mask && (floorf(c / spacing) < (float)Ns)
                          && (floorf(fmodf(c, spacing)) == 0.0f);
    const float s = (full || res) ? 1.0f : 0.0f;
    tr[(size_t)t * (BATCH * D_IN) + idx] = (_Float16)s;
  }
}

// ---------------- fused LIF layer (WMMA, dual batch-tile, async LDS staging) ----
// Block = 4 waves (128 threads). Each wave owns a 16-col weight tile and TWO
// 16-row batch tiles (32 batch rows per block): one weight fragment feeds two
// WMMAs. Membranes live in the accumulator fragments across all T cycles.
// Spike A-tiles are double-buffered in LDS via async-copy-to-LDS (ASYNCcnt).
template <int DIN, int DOUT, bool FINAL>
__global__ __launch_bounds__(128) void lif_layer_kernel(
    const _Float16* __restrict__ trainIn,   // (T, B, DIN) fp16 spikes
    const _Float16* __restrict__ Wh,        // (DOUT, DIN) fp16 weights
    const float*    __restrict__ thp,       // scalar threshold
    _Float16*       __restrict__ trainOut,  // (T, B, DOUT) fp16 spikes (if !FINAL)
    float*          __restrict__ rateOut)   // (B, DOUT) f32 rates      (if FINAL)
{
  constexpr int KCH   = 512;                // K-chunk (halfs)
  constexpr int ROWS  = 32;                 // batch rows per block (2 tiles)
  constexpr int CPT   = DIN / KCH;          // chunks per cycle
  constexpr int TOTAL = SIM_T * CPT;
  __shared__ __align__(16) _Float16 As[2][ROWS * KCH];   // 2 x 32KB

  const int lane = threadIdx.x & 31;
  const int wave = threadIdx.x >> 5;        // 0..3
  const int b0 = blockIdx.y * ROWS;
  const int j0 = blockIdx.x * 64 + wave * 16;
  const int m  = lane & 15;                 // A row / B column within tile
  const int hi = lane >> 4;                 // half-wave selector
  const float th = *thp;

  v8f memb0 = {}, memb1 = {};               // membranes for the two batch tiles
  v8f rate0 = {}, rate1 = {};

  auto stage = [&](int bufIdx, int ci) {
    const int t  = ci / CPT;
    const int kc = (ci % CPT) * KCH;
    const _Float16* src = trainIn + ((size_t)t * BATCH + b0) * DIN + kc;
    _Float16* dst = &As[bufIdx][0];
    #pragma unroll
    for (int i = 0; i < (ROWS * KCH / 8) / 128; ++i) {   // 16 x b128 per thread
      const int chunk = threadIdx.x + i * 128;
      const int row   = chunk >> 6;                      // 0..31
      const int col8  = (chunk & 63) << 3;               // 0..504 step 8
#if USE_ASYNC
      __builtin_amdgcn_global_load_async_to_lds_b128(
          (v4i*)(src + (size_t)row * DIN + col8),
          (v4i*)(dst + row * KCH + col8),
          0, 0);
#else
      *(uint4*)(dst + row * KCH + col8) =
          *(const uint4*)(src + (size_t)row * DIN + col8);
#endif
    }
  };

  stage(0, 0);                              // prologue: fill buffer 0
  for (int ci = 0; ci < TOTAL; ++ci) {
#if USE_ASYNC
    __builtin_amdgcn_s_wait_asynccnt(0);    // own async copies landed in LDS
#endif
    __syncthreads();                        // publish; all readers of 'next' done
    if (ci + 1 < TOTAL) stage((ci + 1) & 1, ci + 1);   // overlap next stage

    const _Float16* buf = &As[ci & 1][0];
    const int t  = ci / CPT;
    const int kc = (ci % CPT) * KCH;
    const _Float16* ap0 = buf + m * KCH + hi * 8;          // batch tile 0
    const _Float16* ap1 = buf + (16 + m) * KCH + hi * 8;   // batch tile 1
    const _Float16* wp  = Wh + (size_t)(j0 + m) * DIN + kc + hi * 8;

    #pragma unroll 4
    for (int k0 = 0; k0 < KCH; k0 += 32) {
      const v8h  bl  = *(const v8h*)(wp + k0);
      const v8h  bh  = *(const v8h*)(wp + k0 + 16);
      const v16h bm  = __builtin_shufflevector(bl, bh, 0, 1, 2, 3, 4, 5, 6, 7, 8, 9, 10, 11, 12, 13, 14, 15);
      const v8h  a0l = *(const v8h*)(ap0 + k0);
      const v8h  a0h = *(const v8h*)(ap0 + k0 + 16);
      const v8h  a1l = *(const v8h*)(ap1 + k0);
      const v8h  a1h = *(const v8h*)(ap1 + k0 + 16);
      v16h a0, a1;
      #pragma unroll
      for (int e = 0; e < 8; ++e) {
        a0[e] = a0l[e]; a0[e + 8] = a0h[e];
        a1[e] = a1l[e]; a1[e + 8] = a1h[e];
      }
      memb0 = __builtin_amdgcn_wmma_f32_16x16x32_f16(
          false, a0, false, bm, (short)0, memb0, false, false);
      memb1 = __builtin_amdgcn_wmma_f32_16x16x32_f16(
          false, a1, false, bm, (short)0, memb1, false, false);
    }

    if ((ci % CPT) == CPT - 1) {            // end of cycle t: LIF fire + reset
      #pragma unroll
      for (int r = 0; r < 8; ++r) {
        const float f0 = (th < memb0[r]) ? 1.0f : 0.0f;
        const float f1 = (th < memb1[r]) ? 1.0f : 0.0f;
        memb0[r] -= th * f0;
        memb1[r] -= th * f1;
        const int M = r + hi * 8;
        if (!FINAL) {
          trainOut[((size_t)t * BATCH + b0 + M)      * DOUT + j0 + m] = (_Float16)f0;
          trainOut[((size_t)t * BATCH + b0 + 16 + M) * DOUT + j0 + m] = (_Float16)f1;
        } else {
          rate0[r] += f0;
          rate1[r] += f1;
        }
      }
    }
  }

  if (FINAL) {
    #pragma unroll
    for (int r = 0; r < 8; ++r) {
      const int M = r + hi * 8;
      rateOut[(size_t)(b0 + M)      * DOUT + j0 + m] = rate0[r] * (1.0f / (float)SIM_T);
      rateOut[(size_t)(b0 + 16 + M) * DOUT + j0 + m] = rate1[r] * (1.0f / (float)SIM_T);
    }
  }
}

// ---------------- launcher ----------------
extern "C" void kernel_launch(void* const* d_in, const int* in_sizes, int n_in,
                              void* d_out, int out_size, void* d_ws, size_t ws_size,
                              hipStream_t stream) {
  const float* x   = (const float*)d_in[0];   // (128, 2048)
  const float* w1  = (const float*)d_in[1];   // (4096, 2048)
  const float* th1 = (const float*)d_in[2];   // scalar
  const float* w2  = (const float*)d_in[3];   // (2048, 4096)
  const float* th2 = (const float*)d_in[4];   // scalar
  float* out = (float*)d_out;                 // (128, 2048)

  char* ws = (char*)d_ws;
  size_t off = 0;
  _Float16* train0 = (_Float16*)(ws + off); off += (size_t)SIM_T * BATCH * D_IN  * 2; // 16MB
  _Float16* train1 = (_Float16*)(ws + off); off += (size_t)SIM_T * BATCH * D_HID * 2; // 32MB
  _Float16* w1h    = (_Float16*)(ws + off); off += (size_t)D_HID * D_IN  * 2;         // 16MB
  _Float16* w2h    = (_Float16*)(ws + off); off += (size_t)D_OUT * D_HID * 2;         // 16MB

  const int nW1 = D_HID * D_IN;
  const int nW2 = D_OUT * D_HID;
  convert_w_kernel<<<(nW1 + 255) / 256, 256, 0, stream>>>(w1, w1h, nW1);
  convert_w_kernel<<<(nW2 + 255) / 256, 256, 0, stream>>>(w2, w2h, nW2);

  encode_kernel<<<(BATCH * D_IN) / 256, 256, 0, stream>>>(x, train0);

  // layer 1: (T,B,2048) -> spikes (T,B,4096)
  lif_layer_kernel<D_IN, D_HID, false>
      <<<dim3(D_HID / 64, BATCH / 32), 128, 0, stream>>>(train0, w1h, th1, train1, nullptr);

  // layer 2: (T,B,4096) -> rates (B,2048)
  lif_layer_kernel<D_HID, D_OUT, true>
      <<<dim3(D_OUT / 64, BATCH / 32), 128, 0, stream>>>(train1, w2h, th2, nullptr, out);
}
